// RLIFLayer_34213709480839
// MI455X (gfx1250) — matrix operand order repaired and checked by reading, not compiled, not run on hardware
//
#include <hip/hip_runtime.h>
#include <hip/hip_bf16.h>
#include <stdint.h>

// ---------------------------------------------------------------------------
// RLIF layer for MI455X (gfx1250):
//   Phase 1: Wx[t][b][h] = sum_i x[b][t][i] * W[h][i]   (bf16 WMMA GEMM)
//   Phase 2: 1000-step recurrent scan, 4 batch-groups x 8 h-slice WGs.
//            V-slice resident in LDS via TDM tensor_load_to_lds; per-step
//            bf16 WMMA st@Vz; cluster barrier + atomic-counter group sync.
// ---------------------------------------------------------------------------

typedef __attribute__((ext_vector_type(16))) __bf16 v16bf;
typedef __attribute__((ext_vector_type(8)))  __bf16 v8bf;
typedef __attribute__((ext_vector_type(4)))  __bf16 v4bf;
typedef __attribute__((ext_vector_type(8)))  float  v8f;
typedef __attribute__((ext_vector_type(4)))  float  v4f;
typedef __attribute__((ext_vector_type(4)))  unsigned int u32x4;
typedef __attribute__((ext_vector_type(8)))  int    i32x8;
typedef __attribute__((ext_vector_type(4)))  int    i32x4;

#define B_   64
#define T_   1000
#define I_   700
#define H_   1024
#define IP_  704          // I padded to multiple of 32
#define KFULL 672         // last guard-free K tile start is 640; 672 is epilogue
#define AMIN 0.81873075307798182f
#define AMAX 0.96078943915232320f

// workspace layout (bytes)
#define WX_BYTES  ((size_t)T_ * B_ * H_ * 4)          // f32  [T][B][H]
#define WT_OFF    (WX_BYTES)
#define WT_BYTES  ((size_t)IP_ * H_ * 2)              // bf16 [IP][H]  (W transposed)
#define VZ_OFF    (WT_OFF + WT_BYTES)
#define VZ_BYTES  ((size_t)H_ * H_ * 2)               // bf16 [H][H]   (V, zero diag)
#define CTR_OFF   ((VZ_OFF + VZ_BYTES + 255) & ~(size_t)255)

// generic -> LDS(addrspace 3) offset (AS3 pointers are 32-bit on amdgcn)
__device__ __forceinline__ unsigned lds_byte_offset(const void* p) {
  return (unsigned)(unsigned long long)
         (__attribute__((address_space(3))) const char*)p;
}

// ---------------------------------------------------------------------------
// Phase 0: transpose/convert W -> Wt bf16 [704][1024], convert V -> Vz bf16
//          with zeroed diagonal, zero the group-barrier counters.
// ---------------------------------------------------------------------------
__global__ void __launch_bounds__(256)
prep_kernel(const float* __restrict__ W, const float* __restrict__ V,
            __bf16* __restrict__ Wt, __bf16* __restrict__ Vz,
            unsigned* __restrict__ ctr) {
  int idx = blockIdx.x * 256 + threadIdx.x;
  if (idx < IP_ * H_) {
    int k = idx / H_, h = idx - k * H_;
    Wt[idx] = (k < I_) ? (__bf16)W[(size_t)h * I_ + k] : (__bf16)0.0f;
  }
  if (idx < H_ * H_) {
    int k = idx >> 10, h = idx & (H_ - 1);
    Vz[idx] = (k == h) ? (__bf16)0.0f : (__bf16)V[idx];
  }
  if (idx < 4) ctr[idx] = 0u;
}

// ---------------------------------------------------------------------------
// Phase 1: GEMM.  One wave computes a 16(t) x 64(h) tile of Wx for one b.
// A: 16x32 bf16 tile of x (converted on the fly), B: 32x16 bf16 tiles of Wt.
// K loop split: guard-free k0 in [0,640], guarded epilogue at k0=672.
// ---------------------------------------------------------------------------
__global__ void __launch_bounds__(256)
gemm_kernel(const float* __restrict__ x, const __bf16* __restrict__ Wt,
            float* __restrict__ Wx) {
  const int lane = threadIdx.x & 31;
  const int wave = threadIdx.x >> 5;
  int gid = blockIdx.x * 8 + wave;
  int hg = gid & 15;                 // 16 groups of 64 h-columns
  int b  = (gid >> 4) & 63;
  int tt = gid >> 10;                // 63 t-tiles of 16
  if (tt >= 63) return;
  const int t0 = tt * 16;

  const int m  = lane & 15;                 // A row (t within tile)
  const int kb = (lane < 16) ? 0 : 8;       // A K sub-block per ISA layout
  int tr = t0 + m; if (tr > T_ - 1) tr = T_ - 1;   // clamp tail tile
  const float* xrow = x + ((size_t)b * T_ + tr) * I_;

  v8f acc0 = {}, acc1 = {}, acc2 = {}, acc3 = {};

  // guard-free main loop: all K indices <= 640+31 < 700
  for (int k0 = 0; k0 < KFULL; k0 += 32) {
    const float* ap = xrow + k0 + kb;
    v4f f0 = *(const v4f*)(ap);
    v4f f1 = *(const v4f*)(ap + 4);
    v4f f2 = *(const v4f*)(ap + 16);
    v4f f3 = *(const v4f*)(ap + 20);
    v16bf A;
#pragma unroll
    for (int e = 0; e < 4; ++e) {
      A[e]      = (__bf16)f0[e];
      A[4 + e]  = (__bf16)f1[e];
      A[8 + e]  = (__bf16)f2[e];
      A[12 + e] = (__bf16)f3[e];
    }
    const __bf16* bcol = Wt + (size_t)(k0 + kb) * H_ + hg * 64 + (lane & 15);
    v16bf B0, B1, B2, B3;
#pragma unroll
    for (int e = 0; e < 8; ++e) {
      size_t o = (size_t)e * H_, o2 = (size_t)(16 + e) * H_;
      B0[e] = bcol[o]; B0[8 + e] = bcol[o2];
      B1[e] = bcol[o + 16]; B1[8 + e] = bcol[o2 + 16];
      B2[e] = bcol[o + 32]; B2[8 + e] = bcol[o2 + 32];
      B3[e] = bcol[o + 48]; B3[8 + e] = bcol[o2 + 48];
    }
    acc0 = __builtin_amdgcn_wmma_f32_16x16x32_bf16(false, A, false, B0, (short)0, acc0, false, false);
    acc1 = __builtin_amdgcn_wmma_f32_16x16x32_bf16(false, A, false, B1, (short)0, acc1, false, false);
    acc2 = __builtin_amdgcn_wmma_f32_16x16x32_bf16(false, A, false, B2, (short)0, acc2, false, false);
    acc3 = __builtin_amdgcn_wmma_f32_16x16x32_bf16(false, A, false, B3, (short)0, acc3, false, false);
  }

  // guarded epilogue: k0 = 672, K indices 672..703, valid while < 700
  {
    const int k0 = KFULL;
    v16bf A;
#pragma unroll
    for (int e = 0; e < 8; ++e) {
      int k = k0 + kb + e;
      A[e] = (k < I_) ? (__bf16)xrow[k] : (__bf16)0.0f;
    }
#pragma unroll
    for (int e = 0; e < 8; ++e) {
      int k = k0 + kb + 16 + e;
      A[8 + e] = (k < I_) ? (__bf16)xrow[k] : (__bf16)0.0f;
    }
    const __bf16* bcol = Wt + (size_t)(k0 + kb) * H_ + hg * 64 + (lane & 15);
    v16bf B0, B1, B2, B3;
#pragma unroll
    for (int e = 0; e < 8; ++e) {
      size_t o = (size_t)e * H_, o2 = (size_t)(16 + e) * H_;
      B0[e] = bcol[o]; B0[8 + e] = bcol[o2];
      B1[e] = bcol[o + 16]; B1[8 + e] = bcol[o2 + 16];
      B2[e] = bcol[o + 32]; B2[8 + e] = bcol[o2 + 32];
      B3[e] = bcol[o + 48]; B3[8 + e] = bcol[o2 + 48];
    }
    acc0 = __builtin_amdgcn_wmma_f32_16x16x32_bf16(false, A, false, B0, (short)0, acc0, false, false);
    acc1 = __builtin_amdgcn_wmma_f32_16x16x32_bf16(false, A, false, B1, (short)0, acc1, false, false);
    acc2 = __builtin_amdgcn_wmma_f32_16x16x32_bf16(false, A, false, B2, (short)0, acc2, false, false);
    acc3 = __builtin_amdgcn_wmma_f32_16x16x32_bf16(false, A, false, B3, (short)0, acc3, false, false);
  }

  const int Mhi = (lane >> 4) * 8;
  const int n   = lane & 15;
#pragma unroll
  for (int j = 0; j < 8; ++j) {
    int t = t0 + j + Mhi;
    if (t < T_) {
      size_t base = ((size_t)t * B_ + b) * H_ + hg * 64 + n;
      Wx[base     ] = acc0[j];
      Wx[base + 16] = acc1[j];
      Wx[base + 32] = acc2[j];
      Wx[base + 48] = acc3[j];
    }
  }
}

// ---------------------------------------------------------------------------
// Phase 2: recurrent scan.  grid = 32 WGs: blockIdx = g*8 + slice.
//   group g owns batch rows 16g..16g+15, slice owns h columns slice*128..+127
//   LDS: sV  = Vz[:, h-slice]  1024x128 bf16 (256 KB, loaded once via TDM)
//        sSt = st_{t-1}        16x1024  bf16 ( 32 KB)
//   d_out doubles as the cross-WG st exchange buffer (each address is
//   written once then read once -> no stale-cache hazard).
// ---------------------------------------------------------------------------
__global__ void __launch_bounds__(256)
scan_kernel(const __bf16* __restrict__ Vz, const float* __restrict__ Wx,
            const float* __restrict__ alpha, const float* __restrict__ ut0,
            const float* __restrict__ st0, float* __restrict__ out,
            unsigned* __restrict__ ctr) {
  __shared__ __align__(16) __bf16 sV[H_ * 128];   // 256 KB
  __shared__ __align__(16) __bf16 sSt[16 * H_];   //  32 KB

  const int lane  = threadIdx.x & 31;
  const int wave  = threadIdx.x >> 5;
  const int g     = blockIdx.x >> 3;      // batch group 0..3
  const int slice = blockIdx.x & 7;       // h slice 0..7
  const int h0    = slice * 128;

  // --- TDM: DMA the 1024x128 bf16 V-slice into LDS (one instruction) ---
  if (wave == 0) {
    unsigned lds = lds_byte_offset(sV);
    uint64_t ga  = (uint64_t)(uintptr_t)(Vz + h0);   // tile start, bytes
    u32x4 g0;
    g0[0] = 1u;                                        // count=1 (valid user D#)
    g0[1] = lds;                                       // lds_addr
    g0[2] = (unsigned)(ga & 0xFFFFFFFFu);              // global_addr[31:0]
    g0[3] = (unsigned)((ga >> 32) & 0x01FFFFFFu) | (2u << 30); // addr[56:32]|type=2
    i32x8 g1;
    g1[0] = (int)(1u << 16);        // data_size = 1 (2 bytes)
    g1[1] = (int)(H_ << 16);        // tensor_dim0[15:0] = 1024
    g1[2] = (int)(H_ << 16);        // tensor_dim1[15:0] = 1024 (rows)
    g1[3] = (int)(128 << 16);       // tile_dim0 = 128 cols
    g1[4] = (int)H_;                // tile_dim1 = 1024 rows
    g1[5] = (int)H_;                // tensor_dim0_stride = 1024 elements
    g1[6] = 0;
    g1[7] = 0;
    i32x4 z4 = {0, 0, 0, 0};
    i32x8 z8 = {0, 0, 0, 0, 0, 0, 0, 0};
    __builtin_amdgcn_tensor_load_to_lds(g0, g1, z4, z4, z8, 0);
    __builtin_amdgcn_s_wait_tensorcnt(0);
  }

  // init sSt = st0 rows of this batch group (float4 -> 4x bf16 packed)
  for (int u = threadIdx.x; u < 16 * (H_ / 4); u += 256) {
    int mm = u >> 8;                  // H_/4 = 256 vec4 per row
    int hh = (u & 255) * 4;
    v4f f = *(const v4f*)(st0 + (size_t)(16 * g + mm) * H_ + hh);
    v4bf p;
    p[0] = (__bf16)f[0]; p[1] = (__bf16)f[1];
    p[2] = (__bf16)f[2]; p[3] = (__bf16)f[3];
    *(v4bf*)(sSt + mm * H_ + hh) = p;
  }
  __syncthreads();

  const int n = h0 + wave * 16 + (lane & 15);   // my output column h
  float a = alpha[n];
  a = fminf(fmaxf(a, AMIN), AMAX);
  const float bet = 1.0f - a;
  const int Mhi = (lane >> 4) * 8;
  const int kb  = (lane < 16) ? 0 : 8;
  const int mA  = lane & 15;

  float ut[8];
#pragma unroll
  for (int j = 0; j < 8; ++j)
    ut[j] = ut0[(size_t)(16 * g + j + Mhi) * H_ + n];

  unsigned target = 0;
  for (int t = 0; t < T_; ++t) {
    // P = st_{t-1} @ Vz[:, my 16 cols]   (M=16, N=16, K=1024)
    v8f acc = {};
    for (int k0 = 0; k0 < H_; k0 += 32) {
      v8bf alo = *(const v8bf*)(sSt + mA * H_ + k0 + kb);
      v8bf ahi = *(const v8bf*)(sSt + mA * H_ + k0 + kb + 16);
      v16bf A  = __builtin_shufflevector(alo, ahi,
                   0,1,2,3,4,5,6,7,8,9,10,11,12,13,14,15);
      v16bf Bv;
#pragma unroll
      for (int e = 0; e < 8; ++e) {
        Bv[e]     = sV[(size_t)(k0 + kb + e) * 128 + wave * 16 + mA];
        Bv[8 + e] = sV[(size_t)(k0 + kb + 16 + e) * 128 + wave * 16 + mA];
      }
      acc = __builtin_amdgcn_wmma_f32_16x16x32_bf16(false, A, false, Bv,
                                                    (short)0, acc, false, false);
    }

    // ut = a*(ut - st_prev) + beta*(wx_t + P);  st = (ut > 1)
#pragma unroll
    for (int j = 0; j < 8; ++j) {
      int M = j + Mhi;
      int b = 16 * g + M;
      float wx = Wx[((size_t)t * B_ + b) * H_ + n];
      float sp = (float)sSt[M * H_ + n];
      float u  = a * (ut[j] - sp) + bet * (wx + acc[j]);
      ut[j] = u;
      out[((size_t)b * T_ + t) * H_ + n] = (u > 1.0f) ? 1.0f : 0.0f;
      if (t + 1 < T_)
        __builtin_prefetch(&Wx[((size_t)(t + 1) * B_ + b) * H_ + n], 0, 1);
    }

    // make spike stores visible, then sync the 8 WGs of this batch group
    __threadfence();
    __syncthreads();
    if (threadIdx.x == 0)
      __hip_atomic_fetch_add(&ctr[g], 1u, __ATOMIC_RELEASE,
                             __HIP_MEMORY_SCOPE_AGENT);
    // real s_barrier_signal/-wait -3 when dispatched with 8-WG clusters;
    // NOP otherwise (ISA: ClusterID==0) -> counter below covers that case.
    __builtin_amdgcn_s_cluster_barrier();
    target += 8;
    if (threadIdx.x == 0) {
      while (__hip_atomic_load(&ctr[g], __ATOMIC_ACQUIRE,
                               __HIP_MEMORY_SCOPE_AGENT) < target)
        __builtin_amdgcn_s_sleep(1);
    }
    __syncthreads();

    // restage full st_t (16 x 1024) into LDS as bf16 (vectorized)
    for (int u = threadIdx.x; u < 16 * (H_ / 4); u += 256) {
      int mm = u >> 8;
      int hh = (u & 255) * 4;
      v4f f = *(const v4f*)(out + ((size_t)(16 * g + mm) * T_ + t) * H_ + hh);
      v4bf p;
      p[0] = (__bf16)f[0]; p[1] = (__bf16)f[1];
      p[2] = (__bf16)f[2]; p[3] = (__bf16)f[3];
      *(v4bf*)(sSt + mm * H_ + hh) = p;
    }
    __syncthreads();
  }
}

// ---------------------------------------------------------------------------
extern "C" void kernel_launch(void* const* d_in, const int* in_sizes, int n_in,
                              void* d_out, int out_size, void* d_ws, size_t ws_size,
                              hipStream_t stream) {
  (void)in_sizes; (void)n_in; (void)out_size; (void)ws_size;
  const float* x     = (const float*)d_in[0];   // [64,1000,700]
  const float* W     = (const float*)d_in[1];   // [1024,700]
  const float* V     = (const float*)d_in[2];   // [1024,1024]
  const float* alpha = (const float*)d_in[3];   // [1024]
  const float* ut0   = (const float*)d_in[4];   // [64,1024]
  const float* st0   = (const float*)d_in[5];   // [64,1024]
  float* out = (float*)d_out;                   // [64,1000,1024]

  char* ws = (char*)d_ws;
  float*    Wx  = (float*)ws;
  __bf16*   Wt  = (__bf16*)(ws + WT_OFF);
  __bf16*   Vz  = (__bf16*)(ws + VZ_OFF);
  unsigned* ctr = (unsigned*)(ws + CTR_OFF);

  prep_kernel<<<4096, 256, 0, stream>>>(W, V, Wt, Vz, ctr);
  gemm_kernel<<<8064, 256, 0, stream>>>(x, Wt, Wx);
  scan_kernel<<<32,   256, 0, stream>>>(Vz, Wx, alpha, ut0, st0, out, ctr);
}
